// InterpretableMultiHeadAttention_50156628083028
// MI455X (gfx1250) — matrix operand order repaired and checked by reading, not compiled, not run on hardware
//
#include <hip/hip_runtime.h>
#include <hip/hip_fp16.h>

typedef _Float16 h8  __attribute__((ext_vector_type(8)));
typedef _Float16 h16 __attribute__((ext_vector_type(16)));
typedef float    f8  __attribute__((ext_vector_type(8)));

#define B_    8
#define T_    2048
#define D_    256
#define DK_   64
#define NROW  (B_ * T_)
#define WAVES 4

// ---------------------------------------------------------------------------
// Butterfly reductions within each 16-lane group (WMMA C-layout row group).
// Prefer v_permlane16_b32 (VALU, no LDS port) over ds_bpermute.
// ---------------------------------------------------------------------------
#if __has_builtin(__builtin_amdgcn_permlane16)
__device__ __forceinline__ float bfly16(float v, unsigned s0, unsigned s1) {
    return __int_as_float(__builtin_amdgcn_permlane16(
        __float_as_int(v), __float_as_int(v), (int)s0, (int)s1, true, false));
}
__device__ __forceinline__ float rowmax16(float v) {
    v = fmaxf(v, bfly16(v, 0x67452301u, 0xEFCDAB89u));   // xor 1
    v = fmaxf(v, bfly16(v, 0x54761032u, 0xDCFE98BAu));   // xor 2
    v = fmaxf(v, bfly16(v, 0x32107654u, 0xBA98FEDCu));   // xor 4
    v = fmaxf(v, bfly16(v, 0xFEDCBA98u, 0x76543210u));   // xor 8
    return v;
}
__device__ __forceinline__ float rowsum16(float v) {
    v += bfly16(v, 0x67452301u, 0xEFCDAB89u);
    v += bfly16(v, 0x54761032u, 0xDCFE98BAu);
    v += bfly16(v, 0x32107654u, 0xBA98FEDCu);
    v += bfly16(v, 0xFEDCBA98u, 0x76543210u);
    return v;
}
#else
__device__ __forceinline__ float rowmax16(float v) {
    v = fmaxf(v, __shfl_xor(v, 1, 32));
    v = fmaxf(v, __shfl_xor(v, 2, 32));
    v = fmaxf(v, __shfl_xor(v, 4, 32));
    v = fmaxf(v, __shfl_xor(v, 8, 32));
    return v;
}
__device__ __forceinline__ float rowsum16(float v) {
    v += __shfl_xor(v, 1, 32);
    v += __shfl_xor(v, 2, 32);
    v += __shfl_xor(v, 4, 32);
    v += __shfl_xor(v, 8, 32);
    return v;
}
#endif

// ---------------------------------------------------------------------------
// Kernel 1: projections  qs = q@Wq, ks = k@Wk, vs = v@Wv  (fp32 -> f16)
// ---------------------------------------------------------------------------
__global__ __launch_bounds__(256) void proj_kernel(
    const float* __restrict__ q,  const float* __restrict__ k,
    const float* __restrict__ v,
    const float* __restrict__ Wq, const float* __restrict__ Wk,
    const float* __restrict__ Wv,
    _Float16* __restrict__ qs, _Float16* __restrict__ ks,
    _Float16* __restrict__ vs)
{
    const int blocksPerTensor = NROW / 16;
    const int which   = blockIdx.x / blocksPerTensor;
    const int rowBase = (blockIdx.x % blocksPerTensor) * 16;

    const float* X = (which == 0) ? q  : (which == 1) ? k  : v;
    const float* W = (which == 0) ? Wq : (which == 1) ? Wk : Wv;
    _Float16*    O = (which == 0) ? qs : (which == 1) ? ks : vs;

    __shared__ float sX[16 * D_];
    const int t = threadIdx.x;
#pragma unroll
    for (int i = 0; i < 16; ++i)
        sX[i * D_ + t] = X[(size_t)(rowBase + i) * D_ + t];
    __syncthreads();

#pragma unroll
    for (int o = 0; o < 4; ++o) {
        const int idx = o * 256 + t;
        const int row = idx >> 6;
        const int col = idx & 63;
        float acc = 0.0f;
        for (int d = 0; d < D_; ++d)
            acc = fmaf(sX[row * D_ + d], W[d * DK_ + col], acc);
        O[(size_t)(rowBase + row) * DK_ + col] = (_Float16)acc;
    }
}

// ---------------------------------------------------------------------------
// Kernel 2: attention.  4 waves per block, one 16-query tile per block.
// Each wave owns an interleaved quarter of the key range with private LDS
// slices (K tile, V^T tile, P tile), so the hot loops need no block barriers.
//   pass 1: S = Q K^T via WMMA, per-wave online row max / sum-exp
//           (merged across waves through LDS once)
//   pass 2: recompute S, p = exp(s-m)/l, write 4 attn copies (non-temporal),
//           accumulate partial head += P @ V via WMMA, merge via ds_add_f32,
//           then out = head @ Wo.
// ---------------------------------------------------------------------------
__global__ __launch_bounds__(32 * WAVES) void attn_kernel(
    const _Float16* __restrict__ qs, const _Float16* __restrict__ ks,
    const _Float16* __restrict__ vs, const float* __restrict__ Wo,
    float* __restrict__ out)
{
    const int tileId = blockIdx.x;               // 0 .. B_*T_/16 - 1
    const int b      = tileId / (T_ / 16);
    const int qBase  = (tileId % (T_ / 16)) * 16;
    const int tid    = threadIdx.x;
    const int wave   = tid >> 5;
    const int lane   = tid & 31;
    const int half   = (lane >> 4) & 1;
    const int nidx   = lane & 15;

    __shared__ alignas(32) _Float16 sQ[16 * DK_];
    __shared__ alignas(32) _Float16 sK [WAVES][32 * DK_];
    __shared__ alignas(32) _Float16 sVt[WAVES][DK_ * 32];   // [dim][key]
    __shared__ alignas(32) _Float16 sP [WAVES][16 * 32];
    __shared__ float sMax[WAVES * 16];
    __shared__ float sSum[WAVES * 16];
    __shared__ alignas(16) float sH[16 * DK_];

    _Float16* sKw  = sK[wave];
    _Float16* sVtw = sVt[wave];
    _Float16* sPw  = sP[wave];

    // ---- load Q tile (16 rows x 64) and zero the head accumulator ----
    if (tid < 16) {
        const _Float16* src = qs + (size_t)(b * T_ + qBase + tid) * DK_;
#pragma unroll
        for (int j = 0; j < DK_; j += 8)
            *(h8*)&sQ[tid * DK_ + j] = *(const h8*)&src[j];
    }
#pragma unroll
    for (int i = tid; i < 16 * DK_; i += 32 * WAVES) sH[i] = 0.0f;
    __syncthreads();

    // ---- build A fragments of Q for k-steps 0 and 32 ----
    // A elem i: K = k0 + i + 16*half, M = nidx
    h16 aQ[2];
#pragma unroll
    for (int s = 0; s < 2; ++s) {
        const _Float16* r = &sQ[nidx * DK_ + s * 32];
        h8 lo = *(const h8*)&r[16 * half];
        h8 hi = *(const h8*)&r[16 * half + 8];
#pragma unroll
        for (int i = 0; i < 8; ++i) { aQ[s][i] = lo[i]; aQ[s][8 + i] = hi[i]; }
    }

    float mrow[8], lrow[8];
#pragma unroll
    for (int r = 0; r < 8; ++r) { mrow[r] = -1e30f; lrow[r] = 0.0f; }

    const float scale = 0.125f;                  // 1/sqrt(64)
    const _Float16* kbase = ks + (size_t)b * T_ * DK_;
    const _Float16* vbase = vs + (size_t)b * T_ * DK_;

    // ================= PASS 1: per-wave running max & sum =================
    for (int kb = wave * 32; kb < T_; kb += 32 * WAVES) {
        const _Float16* srcK = kbase + (size_t)(kb + lane) * DK_;
#pragma unroll
        for (int j = 0; j < DK_; j += 8)
            *(h8*)&sKw[lane * DK_ + j] = *(const h8*)&srcK[j];
        __builtin_amdgcn_wave_barrier();

        f8 ct[2];
#pragma unroll
        for (int nt = 0; nt < 2; ++nt) {
            f8 c = {};
#pragma unroll
            for (int s = 0; s < 2; ++s) {
                // B elem i: K = s*32 + i + 16*half, N(key) = nt*16 + nidx
                h16 bK = *(const h16*)&sKw[(nt * 16 + nidx) * DK_ + s * 32 + 16 * half];
                c = __builtin_amdgcn_wmma_f32_16x16x32_f16(
                        false, aQ[s], false, bK, (short)0, c, false, false);
            }
            ct[nt] = c;
        }
#pragma unroll
        for (int r = 0; r < 8; ++r) {
            const float s0 = ct[0][r] * scale;
            const float s1 = ct[1][r] * scale;
            const float mx   = rowmax16(fmaxf(s0, s1));
            const float mnew = fmaxf(mrow[r], mx);
            const float ps   = rowsum16(__expf(s0 - mnew) + __expf(s1 - mnew));
            lrow[r] = lrow[r] * __expf(mrow[r] - mnew) + ps;
            mrow[r] = mnew;
        }
        __builtin_amdgcn_wave_barrier();
    }

    // ---- merge per-wave softmax stats across the 4 waves ----
    if (nidx == 0) {
#pragma unroll
        for (int r = 0; r < 8; ++r) {
            sMax[wave * 16 + r + 8 * half] = mrow[r];
            sSum[wave * 16 + r + 8 * half] = lrow[r];
        }
    }
    __syncthreads();

    float invl[8];
#pragma unroll
    for (int r = 0; r < 8; ++r) {
        const int row = r + 8 * half;
        float M = sMax[row];
        M = fmaxf(M, sMax[16 + row]);
        M = fmaxf(M, sMax[32 + row]);
        M = fmaxf(M, sMax[48 + row]);
        float L = 0.0f;
#pragma unroll
        for (int w = 0; w < WAVES; ++w)
            L += sSum[w * 16 + row] * __expf(sMax[w * 16 + row] - M);
        mrow[r] = M;
        invl[r] = 1.0f / L;
    }

    // ================= PASS 2: write attn, accumulate P@V =================
    f8 acc[4] = {};
    float* attn0 = out + (size_t)B_ * T_ * D_;           // outputs precede attn
    const size_t copyStride = (size_t)B_ * T_ * T_;

    for (int kb = wave * 32; kb < T_; kb += 32 * WAVES) {
        const _Float16* srcK = kbase + (size_t)(kb + lane) * DK_;
#pragma unroll
        for (int j = 0; j < DK_; j += 8)
            *(h8*)&sKw[lane * DK_ + j] = *(const h8*)&srcK[j];
        const _Float16* srcV = vbase + (size_t)(kb + lane) * DK_;
#pragma unroll
        for (int j = 0; j < DK_; ++j)
            sVtw[j * 32 + lane] = srcV[j];                // transpose into LDS
        __builtin_amdgcn_wave_barrier();

#pragma unroll
        for (int nt = 0; nt < 2; ++nt) {
            f8 c = {};
#pragma unroll
            for (int s = 0; s < 2; ++s) {
                h16 bK = *(const h16*)&sKw[(nt * 16 + nidx) * DK_ + s * 32 + 16 * half];
                c = __builtin_amdgcn_wmma_f32_16x16x32_f16(
                        false, aQ[s], false, bK, (short)0, c, false, false);
            }
#pragma unroll
            for (int r = 0; r < 8; ++r) {
                const float p   = __expf(c[r] * scale - mrow[r]) * invl[r];
                const int   row = r + 8 * half;
                const int   col = kb + nt * 16 + nidx;
                const size_t idx = (size_t)b * T_ * T_
                                 + (size_t)(qBase + row) * T_ + col;
                __builtin_nontemporal_store(p, &attn0[idx]);
                __builtin_nontemporal_store(p, &attn0[idx + copyStride]);
                __builtin_nontemporal_store(p, &attn0[idx + 2 * copyStride]);
                __builtin_nontemporal_store(p, &attn0[idx + 3 * copyStride]);
                sPw[row * 32 + nt * 16 + nidx] = (_Float16)p;
            }
        }
        __builtin_amdgcn_wave_barrier();

        // A fragment of P (16 x 32)
        h16 aP;
        {
            const _Float16* r0 = &sPw[nidx * 32];
            h8 lo = *(const h8*)&r0[16 * half];
            h8 hi = *(const h8*)&r0[16 * half + 8];
#pragma unroll
            for (int i = 0; i < 8; ++i) { aP[i] = lo[i]; aP[8 + i] = hi[i]; }
        }
#pragma unroll
        for (int n = 0; n < 4; ++n) {
            // B elem i: K(key) = i + 16*half, N(dim) = n*16 + nidx -> sVt[dim][key]
            h16 bV = *(const h16*)&sVtw[(n * 16 + nidx) * 32 + 16 * half];
            acc[n] = __builtin_amdgcn_wmma_f32_16x16x32_f16(
                         false, aP, false, bV, (short)0, acc[n], false, false);
        }
        __builtin_amdgcn_wave_barrier();
    }

    // ---- merge partial head tiles (16x64) across waves via LDS atomics ----
#pragma unroll
    for (int n = 0; n < 4; ++n)
#pragma unroll
        for (int r = 0; r < 8; ++r)
            atomicAdd(&sH[(r + 8 * half) * DK_ + n * 16 + nidx], acc[n][r]);
    __syncthreads();

    // ---- out = head @ Wo (64 -> 256); 4 rows per wave ----
    float* obase = out + (size_t)(b * T_ + qBase) * D_;
#pragma unroll
    for (int rr = 0; rr < 4; ++rr) {
        const int row = wave * 4 + rr;
        for (int j = lane; j < D_; j += 32) {
            float a = 0.0f;
#pragma unroll
            for (int d = 0; d < DK_; ++d)
                a = fmaf(sH[row * DK_ + d], Wo[d * D_ + j], a);
            obase[(size_t)row * D_ + j] = a;
        }
    }
}

// ---------------------------------------------------------------------------
extern "C" void kernel_launch(void* const* d_in, const int* in_sizes, int n_in,
                              void* d_out, int out_size, void* d_ws, size_t ws_size,
                              hipStream_t stream)
{
    (void)in_sizes; (void)n_in; (void)out_size; (void)ws_size;

    const float* q  = (const float*)d_in[0];
    const float* k  = (const float*)d_in[1];
    const float* v  = (const float*)d_in[2];
    const float* Wq = (const float*)d_in[3];
    const float* Wk = (const float*)d_in[4];
    const float* Wv = (const float*)d_in[5];
    const float* Wo = (const float*)d_in[6];
    float* out = (float*)d_out;

    _Float16* qs = (_Float16*)d_ws;              // 3 x NROW x 64 f16 = 6 MB scratch
    _Float16* ks = qs + (size_t)NROW * DK_;
    _Float16* vs = ks + (size_t)NROW * DK_;

    proj_kernel<<<3 * (NROW / 16), 256, 0, stream>>>(q, k, v, Wq, Wk, Wv, qs, ks, vs);
    attn_kernel<<<NROW / 16, 32 * WAVES, 0, stream>>>(qs, ks, vs, Wo, out);
}
